// SelfAttention_78271484003105
// MI455X (gfx1250) — matrix operand order repaired and checked by reading, not compiled, not run on hardware
//
#include <hip/hip_runtime.h>
#include <hip/hip_bf16.h>
#include <math.h>

typedef __attribute__((ext_vector_type(16))) _Float16 v16h;
typedef __attribute__((ext_vector_type(8)))  _Float16 v8h;
typedef __attribute__((ext_vector_type(4)))  _Float16 v4h;
typedef __attribute__((ext_vector_type(8)))  float    v8f;
typedef __attribute__((ext_vector_type(4)))  unsigned int v4u;
typedef __attribute__((ext_vector_type(8)))  int      v8i;
typedef __attribute__((ext_vector_type(4)))  int      v4i;

union AFrag { v16h v; v8h h[2]; };

__device__ inline v8f wmma_f16(v16h a, v16h b, v8f c) {
    // 8 args: (neg_a, A, neg_b, B, c_mod, C, reuse_a, reuse_b)
    return __builtin_amdgcn_wmma_f32_16x16x32_f16(false, a, false, b, (short)0, c, false, false);
}

// LDS byte offset of a __shared__ object (LDS aperture: addr[31:0] = offset)
__device__ inline unsigned ldsof(const void* p) { return (unsigned)(uintptr_t)p; }

// ---------------------------------------------------------------------------
// TDM: async 2D f16 tile load Global -> LDS (descriptor per ISA 08 §8).
// tileD0 = contiguous elements per line, tileD1 = #lines, strideElems = line
// stride in elements. EXEC ignored; tracked by TENSORcnt.
// ---------------------------------------------------------------------------
__device__ inline void tdm_load_2d_f16(unsigned lds_addr, const void* gptr,
                                       unsigned tileD0, unsigned tileD1,
                                       unsigned long long strideElems) {
    unsigned long long ga = (unsigned long long)(uintptr_t)gptr;
    v4u g0;
    g0[0] = 1u;                                        // count=1 (valid), user desc
    g0[1] = lds_addr;                                  // lds_addr [63:32]
    g0[2] = (unsigned)(ga & 0xffffffffu);              // global_addr [95:64]
    g0[3] = (unsigned)((ga >> 32) & 0x1ffffffu)        // global_addr [120:96]
            | (2u << 30);                              // type=2 ("image")
    const unsigned tensorD0 = (unsigned)strideElems;   // generous OOB dims
    const unsigned tensorD1 = 0x40000000u;
    v8i g1;
    g1[0] = (int)(1u << 16);                                   // data_size=1 (2B)
    g1[1] = (int)((tensorD0 & 0xffffu) << 16);                 // tdim0[15:0]
    g1[2] = (int)((tensorD0 >> 16) | ((tensorD1 & 0xffffu) << 16));
    g1[3] = (int)((tensorD1 >> 16) | (tileD0 << 16));          // tile_dim0
    g1[4] = (int)(tileD1 & 0xffffu);                           // tile_dim1, tile_dim2=0
    g1[5] = (int)(strideElems & 0xffffffffu);                  // dim0_stride[31:0]
    g1[6] = (int)((strideElems >> 32) & 0xffffu);              // dim0_stride[47:32]
    g1[7] = 0;
    v4i g2 = {0, 0, 0, 0};
    v4i g3 = {0, 0, 0, 0};
#if defined(__clang_major__) && __clang_major__ >= 23
    v8i g4 = {0, 0, 0, 0, 0, 0, 0, 0};
    __builtin_amdgcn_tensor_load_to_lds(g0, g1, g2, g3, g4, 0);
#else
    __builtin_amdgcn_tensor_load_to_lds(g0, g1, g2, g3, 0);
#endif
}

// ---------------------------------------------------------------------------
// fp32 -> f16 convert with scale, 4 elements/thread
// ---------------------------------------------------------------------------
__global__ __launch_bounds__(256) void cvt_f32_f16(const float4* __restrict__ in,
                                                   _Float16* __restrict__ out,
                                                   float scale, int n4) {
    int i = blockIdx.x * 256 + threadIdx.x;
    if (i < n4) {
        float4 v = in[i];
        v4h h;
        h[0] = (_Float16)(v.x * scale);
        h[1] = (_Float16)(v.y * scale);
        h[2] = (_Float16)(v.z * scale);
        h[3] = (_Float16)(v.w * scale);
        *(v4h*)(out + 4 * (size_t)i) = h;
    }
}

// ---------------------------------------------------------------------------
// WMMA GEMM: C[M][N] = A[M][K] * B[N][K]^T (both row-major, contraction along
// fast axis -> all fragment loads contiguous). 128 threads = 4 waves (2Mx2N),
// wave tile 64x32, block tile 128x64. M = gridDim.y*128, N = gridDim.x*64.
// Register double-buffered K loop: 8 WMMAs overlap next fragment loads.
// ---------------------------------------------------------------------------
struct Frags { AFrag a[4]; v16h b[2]; };

__device__ inline void load_frags(Frags& f, const _Float16* const* aP,
                                  const _Float16* const* bP, int k) {
#pragma unroll
    for (int i = 0; i < 4; ++i) {
        f.a[i].h[0] = *(const v8h*)(aP[i] + k);
        f.a[i].h[1] = *(const v8h*)(aP[i] + k + 8);
    }
#pragma unroll
    for (int j = 0; j < 2; ++j) f.b[j] = *(const v16h*)(bP[j] + k);
}

__device__ inline void mma_step(v8f (&acc)[4][2], const Frags& f) {
#pragma unroll
    for (int i = 0; i < 4; ++i)
#pragma unroll
        for (int j = 0; j < 2; ++j) acc[i][j] = wmma_f16(f.a[i].v, f.b[j], acc[i][j]);
}

template <bool OUT_F32_BIAS>
__global__ __launch_bounds__(128) void gemm_abt(const _Float16* __restrict__ A,
                                                const _Float16* __restrict__ B,
                                                int K,
                                                _Float16* __restrict__ outH,
                                                float* __restrict__ outF,
                                                const float* __restrict__ bias) {
    const int lane = threadIdx.x & 31;
    const int wave = threadIdx.x >> 5;
    const int N = gridDim.x << 6;
    const int mBase = (blockIdx.y << 7) + ((wave >> 1) << 6);   // 64-row wave tile
    const int nBase = (blockIdx.x << 6) + ((wave & 1) << 5);    // 32-col wave tile

    const int koffA = (lane >> 4) << 4;
    const int koffB = (lane >> 4) << 4;

    const _Float16* aP[4];
    const _Float16* bP[2];
#pragma unroll
    for (int i = 0; i < 4; ++i)
        aP[i] = A + (size_t)(mBase + i * 16 + (lane & 15)) * K + koffA;
#pragma unroll
    for (int j = 0; j < 2; ++j)
        bP[j] = B + (size_t)(nBase + j * 16 + (lane & 15)) * K + koffB;

    v8f acc[4][2] = {};
    Frags f0, f1;
    load_frags(f0, aP, bP, 0);
    for (int k = 0; k + 64 <= K; k += 64) {
        __builtin_prefetch(aP[0] + k + 512, 0, 1);       // global_prefetch_b8
        load_frags(f1, aP, bP, k + 32);
        mma_step(acc, f0);
        if (k + 64 < K) load_frags(f0, aP, bP, k + 64);
        mma_step(acc, f1);
    }

    const int nOut = lane & 15;
    const int mOff = (lane >> 4) << 3;                   // C layout half-lane rows
#pragma unroll
    for (int i = 0; i < 4; ++i)
#pragma unroll
        for (int j = 0; j < 2; ++j) {
            const int mG = mBase + i * 16 + mOff;
            const int nG = nBase + j * 16 + nOut;
            if (OUT_F32_BIAS) {
                float bv = bias[nG];
#pragma unroll
                for (int r = 0; r < 8; ++r)
                    outF[(size_t)(mG + r) * N + nG] = acc[i][j][r] + bv;
            } else {
#pragma unroll
                for (int r = 0; r < 8; ++r)
                    outH[(size_t)(mG + r) * N + nG] = (_Float16)acc[i][j][r];
            }
        }
}

// ---------------------------------------------------------------------------
// Flash attention: grid (b*h=64, t/64=16), 128 threads = 4 waves, each wave
// owns 16 q-rows. K/V 32-key tiles staged into double-buffered LDS by the
// Tensor Data Mover (wave 0 issues next-block descriptors at loop top; waits
// TENSORcnt at loop bottom; one workgroup barrier per block). Softmax uses
// the full wave: each row's 32 columns split across the two half-waves,
// combined with __shfl_xor(16) -> EXEC stays all-ones (no saveexec).
// Qh, Kh: [4096][1024] f16 (Q pre-scaled by 1/8). Vt: [1024][4096] f16.
// ---------------------------------------------------------------------------
__global__ __launch_bounds__(128) void attention(const _Float16* __restrict__ Qh,
                                                 const _Float16* __restrict__ Kh,
                                                 const _Float16* __restrict__ Vt,
                                                 _Float16* __restrict__ Oh) {
    __shared__ _Float16 KtS[2][32][64];   // [buf][key][d]    8 KB
    __shared__ _Float16 VtS[2][64][32];   // [buf][d][key]    8 KB
    __shared__ float    Ssm[4][16][32];
    __shared__ _Float16 Psm[4][16][32];
    __shared__ float    Alp[4][16];
    __shared__ float    Lrw[4][16];

    const int lane = threadIdx.x & 31;
    const int wave = threadIdx.x >> 5;
    const int bh = blockIdx.x;              // 0..63
    const int rowBase = (bh >> 4) << 10;    // batch * 1024 tokens
    const int hbase = (bh & 15) << 6;       // head * 64
    const int qBase = (blockIdx.y << 6) + (wave << 4);

    const int m = lane & 15;
    const int half = lane >> 4;
    const int coff = half << 4;             // softmax column split per half-wave

    // Q A-fragments for d-chunks {0..31, 32..63}
    const _Float16* qp = Qh + (size_t)(rowBase + qBase + m) * 1024 + hbase;
    AFrag aq0, aq1;
    aq0.h[0] = *(const v8h*)(qp + (half << 4));
    aq0.h[1] = *(const v8h*)(qp + (half << 4) + 8);
    aq1.h[0] = *(const v8h*)(qp + 32 + (half << 4));
    aq1.h[1] = *(const v8h*)(qp + 32 + (half << 4) + 8);

    // Prologue: TDM-stage key block 0 into buffer 0
    if (wave == 0) {
        tdm_load_2d_f16(ldsof(&KtS[0][0][0]),
                        Kh + (size_t)rowBase * 1024 + hbase, 64, 32, 1024ull);
        tdm_load_2d_f16(ldsof(&VtS[0][0][0]),
                        Vt + (size_t)hbase * 4096 + rowBase, 32, 64, 4096ull);
        __builtin_amdgcn_s_wait_tensorcnt(0);
    }
    __syncthreads();

    v8f o0 = {}, o1 = {}, o2 = {}, o3 = {};
    float mprev = -INFINITY, lsum = 0.0f;   // identical in both half-waves
    int buf = 0;

    for (int kb = 0; kb < 1024; kb += 32) {
        // TDM: prefetch next key block into the other buffer (readers of that
        // buffer all passed the barrier at the end of the previous iteration)
        if (wave == 0 && kb + 32 < 1024) {
            tdm_load_2d_f16(ldsof(&KtS[buf ^ 1][0][0]),
                            Kh + (size_t)(rowBase + kb + 32) * 1024 + hbase,
                            64, 32, 1024ull);
            tdm_load_2d_f16(ldsof(&VtS[buf ^ 1][0][0]),
                            Vt + (size_t)hbase * 4096 + rowBase + kb + 32,
                            32, 64, 4096ull);
        }

        // ---- batch all K B-fragments, then 4 WMMAs (one dscnt wait) ----
        const _Float16* kp0 = &KtS[buf][m][half << 4];
        const _Float16* kp1 = &KtS[buf][16 + m][half << 4];
        v16h bk00 = *(const v16h*)(kp0);        // keys 0..15,  d 0..31 chunk
        v16h bk01 = *(const v16h*)(kp0 + 32);   // keys 0..15,  d 32..63 chunk
        v16h bk10 = *(const v16h*)(kp1);        // keys 16..31, d 0..31 chunk
        v16h bk11 = *(const v16h*)(kp1 + 32);   // keys 16..31, d 32..63 chunk

        v8f s0 = {}, s1 = {};
        s0 = wmma_f16(aq0.v, bk00, s0);
        s0 = wmma_f16(aq1.v, bk01, s0);
        s1 = wmma_f16(aq0.v, bk10, s1);
        s1 = wmma_f16(aq1.v, bk11, s1);
#pragma unroll
        for (int r = 0; r < 8; ++r) {
            Ssm[wave][half * 8 + r][m]      = s0[r];
            Ssm[wave][half * 8 + r][16 + m] = s1[r];
        }

        // ---- issue V fragment loads early: latency hides under softmax ----
        const _Float16* vp0 = &VtS[buf][m][half << 4];
        v16h vb0 = *(const v16h*)(vp0);
        v16h vb1 = *(const v16h*)(vp0 + 16 * 32);
        v16h vb2 = *(const v16h*)(vp0 + 32 * 32);
        v16h vb3 = *(const v16h*)(vp0 + 48 * 32);

        // ---- online softmax, full wave: lane handles row m, cols coff..+15;
        //      halves combined with shfl_xor(16) (both keep identical state) --
        float cmax = Ssm[wave][m][coff];
#pragma unroll
        for (int j = 1; j < 16; ++j) cmax = fmaxf(cmax, Ssm[wave][m][coff + j]);
        cmax = fmaxf(cmax, __shfl_xor(cmax, 16, 32));
        float mnew = fmaxf(mprev, cmax);
        float alpha = __expf(mprev - mnew);     // exp(-inf)=0 on first pass
        float part = 0.0f;
#pragma unroll
        for (int j = 0; j < 16; ++j) {
            float p = __expf(Ssm[wave][m][coff + j] - mnew);
            part += p;
            Psm[wave][m][coff + j] = (_Float16)p;
        }
        lsum = lsum * alpha + part + __shfl_xor(part, 16, 32);
        Alp[wave][m] = alpha;                   // both halves store same value
        mprev = mnew;

        // ---- rescale O accumulators (lane holds rows half*8 .. half*8+7) ----
        float ar[8];
#pragma unroll
        for (int r = 0; r < 8; ++r) ar[r] = Alp[wave][half * 8 + r];
#pragma unroll
        for (int r = 0; r < 8; ++r) { o0[r] *= ar[r]; o1[r] *= ar[r]; o2[r] *= ar[r]; o3[r] *= ar[r]; }

        // ---- P fragment from LDS, then O += P * V ----
        AFrag ap;
        const _Float16* pp = &Psm[wave][m][half << 4];
        ap.h[0] = *(const v8h*)pp;
        ap.h[1] = *(const v8h*)(pp + 8);

        o0 = wmma_f16(ap.v, vb0, o0);
        o1 = wmma_f16(ap.v, vb1, o1);
        o2 = wmma_f16(ap.v, vb2, o2);
        o3 = wmma_f16(ap.v, vb3, o3);

        // ---- publish next buffer: TDM complete + all waves past this block --
        if (wave == 0) __builtin_amdgcn_s_wait_tensorcnt(0);
        __syncthreads();
        buf ^= 1;
    }

    Lrw[wave][m] = lsum;                        // both halves store same value

    float rinv[8];
#pragma unroll
    for (int r = 0; r < 8; ++r) rinv[r] = 1.0f / Lrw[wave][half * 8 + r];

    v8f ot[4] = {o0, o1, o2, o3};
#pragma unroll
    for (int nb = 0; nb < 4; ++nb)
#pragma unroll
        for (int r = 0; r < 8; ++r)
            Oh[(size_t)(rowBase + qBase + half * 8 + r) * 1024 + hbase + nb * 16 + m] =
                (_Float16)(ot[nb][r] * rinv[r]);
}

// ---------------------------------------------------------------------------
extern "C" void kernel_launch(void* const* d_in, const int* in_sizes, int n_in,
                              void* d_out, int out_size, void* d_ws, size_t ws_size,
                              hipStream_t stream) {
    const float* x  = (const float*)d_in[0];
    const float* Wq = (const float*)d_in[1];
    const float* Wk = (const float*)d_in[2];
    const float* Wv = (const float*)d_in[3];
    const float* Wo = (const float*)d_in[4];
    const float* bo = (const float*)d_in[5];

    char* ws = (char*)d_ws;
    const size_t MB = 1u << 20;
    _Float16* Xh  = (_Float16*)(ws + 0 * MB);    // 4096x1024  (8 MB)
    _Float16* Wqh = (_Float16*)(ws + 8 * MB);    // 1024x1024  (2 MB)
    _Float16* Wkh = (_Float16*)(ws + 10 * MB);
    _Float16* Wvh = (_Float16*)(ws + 12 * MB);
    _Float16* Woh = (_Float16*)(ws + 14 * MB);
    _Float16* Qh  = (_Float16*)(ws + 16 * MB);   // 4096x1024
    _Float16* Kh  = (_Float16*)(ws + 24 * MB);   // 4096x1024
    _Float16* Vt  = (_Float16*)(ws + 32 * MB);   // 1024x4096 (transposed V)
    _Float16* Oh  = (_Float16*)(ws + 40 * MB);   // 4096x1024

    const int nX4 = 4096 * 1024 / 4, nW4 = 1024 * 1024 / 4;
    cvt_f32_f16<<<nX4 / 256, 256, 0, stream>>>((const float4*)x,  Xh,  1.0f,   nX4);
    cvt_f32_f16<<<nW4 / 256, 256, 0, stream>>>((const float4*)Wq, Wqh, 0.125f, nW4);
    cvt_f32_f16<<<nW4 / 256, 256, 0, stream>>>((const float4*)Wk, Wkh, 1.0f,   nW4);
    cvt_f32_f16<<<nW4 / 256, 256, 0, stream>>>((const float4*)Wv, Wvh, 1.0f,   nW4);
    cvt_f32_f16<<<nW4 / 256, 256, 0, stream>>>((const float4*)Wo, Woh, 1.0f,   nW4);

    dim3 gQK(1024 / 64, 4096 / 128);  // (N/64, M/128)
    gemm_abt<false><<<gQK, 128, 0, stream>>>(Xh, Wqh, 1024, Qh, nullptr, nullptr);
    gemm_abt<false><<<gQK, 128, 0, stream>>>(Xh, Wkh, 1024, Kh, nullptr, nullptr);

    dim3 gV(4096 / 64, 1024 / 128);   // transposed output: M=features, N=tokens
    gemm_abt<false><<<gV, 128, 0, stream>>>(Wvh, Xh, 1024, Vt, nullptr, nullptr);

    dim3 gA(64, 16);                  // (b*h, t/64)
    attention<<<gA, 128, 0, stream>>>(Qh, Kh, Vt, Oh);

    gemm_abt<true><<<gQK, 128, 0, stream>>>(Oh, Woh, 1024, nullptr, (float*)d_out, bo);
}
